// ConvLinearAttention_88596585382405
// MI455X (gfx1250) — compile-verified, matrix operand order checked by reading, not run on hardware
//
#include <hip/hip_runtime.h>

#define B_     16
#define C_     256
#define N_     4096
#define O3_    384
#define INNER_ 128
#define HEADS_ 4
#define DH_    32
#define SCALE_ 0.1767766952966369f
#define EPS_   1e-5f

typedef __attribute__((ext_vector_type(16))) __bf16 v16bf;
typedef __attribute__((ext_vector_type(8)))  __bf16 v8bf;
typedef __attribute__((ext_vector_type(8)))  float  v8f;
typedef unsigned int u32x4 __attribute__((ext_vector_type(4)));
typedef int          i32x4 __attribute__((ext_vector_type(4)));
typedef int          i32x8 __attribute__((ext_vector_type(8)));

#if defined(__HIP_DEVICE_COMPILE__) && __has_builtin(__builtin_amdgcn_tensor_load_to_lds)
#define HAVE_TDM 1
#else
#define HAVE_TDM 0
#endif

// 16-bit A/B fragment K-index for lane `lane`, element `e`
// K = (e>>3)*16 + (lane>>4)*8 + (e&7)   (ISA 7.12.2)
__device__ __forceinline__ int kmap16(int lane, int e) {
  return ((e >> 3) << 4) | ((lane >> 4) << 3) | (e & 7);
}

__device__ __forceinline__ v8f wmma_bf16(v16bf a, v16bf b, v8f c) {
  return __builtin_amdgcn_wmma_f32_16x16x32_bf16(false, a, false, b, (short)0, c,
                                                 false, false);
}

// Load one 16-element bf16 fragment: per-lane K chunks are contiguous, so this
// is two global_load_b128s.  p must point at K = k0 + (lane>>4)*8.
__device__ __forceinline__ v16bf frag_ld(const __bf16* p) {
  v8bf lo = *(const v8bf*)p;
  v8bf hi = *(const v8bf*)(p + 16);
  return __builtin_shufflevector(lo, hi, 0, 1, 2, 3, 4, 5, 6, 7,
                                 8, 9, 10, 11, 12, 13, 14, 15);
}

#if HAVE_TDM
// TDM: DMA a [tile_k x tile_n] fp32 tile (row stride = stride_e elements) into LDS.
__device__ __forceinline__ void tdm_load_tile_f32(void* lds_dst, const float* gsrc,
                                                  int tile_n, int tile_k,
                                                  int tensor_n, int tensor_k,
                                                  long long stride_e) {
  unsigned long long ga = (unsigned long long)(const void*)gsrc;
  unsigned lds = (unsigned)(unsigned long long)lds_dst;  // low 32 bits = LDS offset
  u32x4 g0;
  g0.x = 1u;                                              // count=1, user mode
  g0.y = lds;                                             // lds_addr
  g0.z = (unsigned)ga;                                    // global_addr[31:0]
  g0.w = (unsigned)((ga >> 32) & 0x1FFFFFFull) | (2u << 30); // addr[56:32] | type=2
  i32x8 g1;
  g1[0] = (2 << 16);                                      // wg_mask=0, data_size=4B
  g1[1] = (tensor_n & 0xFFFF) << 16;                      // tensor_dim0[15:0]
  g1[2] = ((tensor_n >> 16) & 0xFFFF) | ((tensor_k & 0xFFFF) << 16);
  g1[3] = ((tensor_k >> 16) & 0xFFFF) | (tile_n << 16);   // tile_dim0
  g1[4] = tile_k & 0xFFFF;                                // tile_dim1 (tile_dim2=0)
  g1[5] = (int)(stride_e & 0xFFFFFFFFll);                 // dim0_stride[31:0]
  g1[6] = (int)((stride_e >> 32) & 0xFFFFll);             // dim0_stride[47:32]
  g1[7] = 0;
  i32x4 z4 = {0, 0, 0, 0};
#if __clang_major__ >= 23
  i32x8 z8 = {0, 0, 0, 0, 0, 0, 0, 0};
  __builtin_amdgcn_tensor_load_to_lds(g0, g1, z4, z4, z8, 0);
#else
  __builtin_amdgcn_tensor_load_to_lds(g0, g1, z4, z4, 0);
#endif
}
#endif

// ---------------------------------------------------------------------------
// 0) weights -> bf16.  grid 512, block 256.
// ---------------------------------------------------------------------------
__global__ __launch_bounds__(256) void k_cvt_w(const float* __restrict__ wqkv,
                                               const float* __restrict__ wout,
                                               __bf16* __restrict__ wqb,
                                               __bf16* __restrict__ wob) {
  const int i = blockIdx.x * 256 + threadIdx.x;
  if (i < O3_ * C_) wqb[i] = (__bf16)wqkv[i];
  const int j = i - O3_ * C_;
  if (j >= 0 && j < C_ * INNER_) wob[j] = (__bf16)wout[j];
}

// ---------------------------------------------------------------------------
// 1) qkv GEMM: [384x256] @ [256x4096] per batch.  x tile staged in LDS by TDM.
//    q,k out fp32 (softmax precision); v out bf16 directly.
//    grid (64, 12, 16), block 256 (8 waves; WG = 32o x 64n).
// ---------------------------------------------------------------------------
__global__ __launch_bounds__(256) void k_qkv_gemm(const float* __restrict__ x,
                                                  const __bf16* __restrict__ wqb,
                                                  float* __restrict__ qf,
                                                  float* __restrict__ kf,
                                                  __bf16* __restrict__ vbf) {
  const int lane = threadIdx.x & 31;
  const int wave = threadIdx.x >> 5;
  const int b = blockIdx.z;
  const int oBase = blockIdx.y * 32 + (wave >> 2) * 16;
  const int nWG = blockIdx.x * 64;
  const int nLoc = (wave & 3) * 16;
  const int row = oBase + (lane & 15);
  const int col = nWG + nLoc + (lane & 15);
  const int off = (lane >> 4) << 3;
  const float* xb = x + (size_t)b * C_ * N_;

  __shared__ __align__(16) float xtile[32 * 64];

  v8f acc = {};
  for (int k0 = 0; k0 < C_; k0 += 32) {
    __syncthreads();
#if HAVE_TDM
    if (wave == 0) {
      tdm_load_tile_f32(xtile, xb + (size_t)k0 * N_ + nWG, 64, 32, N_, C_ - k0,
                        (long long)N_);
      __builtin_amdgcn_s_wait_tensorcnt(0);
    }
#else
    for (int i = threadIdx.x; i < 32 * 16; i += 256)
      ((float4*)xtile)[i] =
          *(const float4*)(xb + (size_t)(k0 + (i >> 4)) * N_ + nWG + ((i & 15) << 2));
#endif
    if (k0 + 32 < C_)
      __builtin_prefetch(xb + (size_t)(k0 + 32) * N_ + nWG + lane * 8, 0, 1);
    __syncthreads();

    v16bf a = frag_ld(wqb + row * C_ + k0 + off);
    v16bf bm;
#pragma unroll
    for (int e = 0; e < 16; ++e)
      bm[e] = (__bf16)xtile[kmap16(lane, e) * 64 + nLoc + (lane & 15)];
    acc = wmma_bf16(a, bm, acc);
  }
#pragma unroll
  for (int r = 0; r < 8; ++r) {
    const int m = oBase + r + ((lane >> 4) << 3);
    if (m < INNER_)
      qf[(size_t)(b * INNER_ + m) * N_ + col] = acc[r];
    else if (m < 2 * INNER_)
      kf[(size_t)(b * INNER_ + m - INNER_) * N_ + col] = acc[r];
    else
      vbf[(size_t)(b * INNER_ + m - 2 * INNER_) * N_ + col] = (__bf16)acc[r];
  }
}

// ---------------------------------------------------------------------------
// 2) softmax of k over n; writes bf16 IN PLACE over its own fp32 row
//    (values cached in registers).  grid (32, 4, 16), block 256.
// ---------------------------------------------------------------------------
__global__ __launch_bounds__(256) void k_ksoftmax(float* __restrict__ kf) {
  const int d = blockIdx.x, h = blockIdx.y, b = blockIdx.z;
  float* row = kf + (size_t)(b * INNER_ + h * DH_ + d) * N_;
  __shared__ float s[256];
  const int t = threadIdx.x;
  float vv[16];
  float m = -3.4e38f;
#pragma unroll
  for (int j = 0; j < 16; ++j) {
    vv[j] = row[t * 16 + j];
    m = fmaxf(m, vv[j]);
  }
  s[t] = m; __syncthreads();
  for (int o = 128; o > 0; o >>= 1) {
    if (t < o) s[t] = fmaxf(s[t], s[t + o]);
    __syncthreads();
  }
  m = s[0]; __syncthreads();
  float sum = 0.f;
#pragma unroll
  for (int j = 0; j < 16; ++j) { vv[j] = __expf(vv[j] - m); sum += vv[j]; }
  s[t] = sum; __syncthreads();
  for (int o = 128; o > 0; o >>= 1) {
    if (t < o) s[t] += s[t + o];
    __syncthreads();
  }
  const float inv = 1.0f / s[0];
  __bf16* rowb = (__bf16*)row;                // bf16 row in first half of fp32 row
  v8bf pk[2];
#pragma unroll
  for (int j = 0; j < 16; ++j) pk[j >> 3][j & 7] = (__bf16)(vv[j] * inv);
  *(v8bf*)(rowb + t * 16) = pk[0];
  *(v8bf*)(rowb + t * 16 + 8) = pk[1];
}

// ---------------------------------------------------------------------------
// 3) softmax of q over d (*SCALE); writes transposed bf16 q_t[(b,h)][n][32].
//    grid (16, 4, 16), block 256.
// ---------------------------------------------------------------------------
__global__ __launch_bounds__(256) void k_qsoftmax(const float* __restrict__ qf,
                                                  __bf16* __restrict__ qbt) {
  const int n = blockIdx.x * 256 + threadIdx.x;
  const int h = blockIdx.y, b = blockIdx.z;
  const float* base = qf + (size_t)(b * INNER_ + h * DH_) * N_ + n;
  float v[DH_];
  float m = -3.4e38f;
#pragma unroll
  for (int d = 0; d < DH_; ++d) { v[d] = base[(size_t)d * N_]; m = fmaxf(m, v[d]); }
  float sum = 0.f;
#pragma unroll
  for (int d = 0; d < DH_; ++d) { v[d] = __expf(v[d] - m); sum += v[d]; }
  const float sc = SCALE_ / sum;
  v8bf pk[4];
#pragma unroll
  for (int d = 0; d < DH_; ++d) pk[d >> 3][d & 7] = (__bf16)(v[d] * sc);
  v8bf* dst = (v8bf*)(qbt + ((size_t)(b * HEADS_ + h) * N_ + n) * DH_);
#pragma unroll
  for (int j = 0; j < 4; ++j) dst[j] = pk[j];
}

// ---------------------------------------------------------------------------
// 4) context[d][e] = sum_n k'[d][n] v[e][n] per (b,h).  grid 64, block 256.
//    bf16 b128 fragment loads; 8 waves split K=4096; ds_add_f32 reduce.
// ---------------------------------------------------------------------------
__global__ __launch_bounds__(256) void k_context(const float* __restrict__ kf,
                                                 const __bf16* __restrict__ vbf,
                                                 float* __restrict__ ctx) {
  const int bh = blockIdx.x;
  const int b = bh >> 2, h = bh & 3;
  const int lane = threadIdx.x & 31;
  const int wave = threadIdx.x >> 5;
  const int off = (lane >> 4) << 3;
  const float* kbase = kf + (size_t)(b * INNER_ + h * DH_) * N_;
  const __bf16* kr0 = (const __bf16*)(kbase + (size_t)(lane & 15) * N_);
  const __bf16* kr1 = (const __bf16*)(kbase + (size_t)(16 + (lane & 15)) * N_);
  const __bf16* vr0 = vbf + (size_t)(b * INNER_ + h * DH_ + (lane & 15)) * N_;
  const __bf16* vr1 = vr0 + (size_t)16 * N_;

  __shared__ float ctxs[DH_ * DH_];
  for (int i = threadIdx.x; i < DH_ * DH_; i += 256) ctxs[i] = 0.f;
  __syncthreads();

  v8f acc[2][2] = {};
  const int kBeg = wave * 512, kEnd = kBeg + 512;
  for (int k0 = kBeg; k0 < kEnd; k0 += 32) {
    v16bf a0 = frag_ld(kr0 + k0 + off);
    v16bf a1 = frag_ld(kr1 + k0 + off);
    v16bf b0 = frag_ld(vr0 + k0 + off);
    v16bf b1 = frag_ld(vr1 + k0 + off);
    acc[0][0] = wmma_bf16(a0, b0, acc[0][0]);
    acc[0][1] = wmma_bf16(a0, b1, acc[0][1]);
    acc[1][0] = wmma_bf16(a1, b0, acc[1][0]);
    acc[1][1] = wmma_bf16(a1, b1, acc[1][1]);
  }
#pragma unroll
  for (int i = 0; i < 2; ++i)
#pragma unroll
    for (int j = 0; j < 2; ++j)
#pragma unroll
      for (int r = 0; r < 8; ++r) {
        const int m = i * 16 + r + ((lane >> 4) << 3);
        const int e = j * 16 + (lane & 15);
        atomicAdd(&ctxs[m * DH_ + e], acc[i][j][r]);
      }
  __syncthreads();
  for (int i = threadIdx.x; i < DH_ * DH_; i += 256)
    ctx[(size_t)bh * DH_ * DH_ + i] = ctxs[i];
}

// ---------------------------------------------------------------------------
// 5) attn[e][n] = sum_d ctx[d][e] q'[d][n]; K=32 -> one WMMA per tile.
//    Writes transposed bf16 attn_t[b][n][128] (8 accs = 8 consecutive channels
//    -> one packed b128 store).  grid (32, 64), block 256.
// ---------------------------------------------------------------------------
__global__ __launch_bounds__(256) void k_attn_out(const __bf16* __restrict__ qbt,
                                                  const float* __restrict__ ctx,
                                                  __bf16* __restrict__ attn_t) {
  const int bh = blockIdx.y;
  const int b = bh >> 2, h = bh & 3;
  const int lane = threadIdx.x & 31;
  const int wave = threadIdx.x >> 5;
  const int nBase = blockIdx.x * 128 + wave * 16;
  const int n = nBase + (lane & 15);
  const int off = (lane >> 4) << 3;

  __shared__ float ctxs[DH_ * DH_];
  for (int i = threadIdx.x; i < DH_ * DH_; i += 256)
    ctxs[i] = ctx[(size_t)bh * DH_ * DH_ + i];
  __syncthreads();

  v16bf bm = frag_ld(qbt + ((size_t)bh * N_ + n) * DH_ + off);

#pragma unroll
  for (int i = 0; i < 2; ++i) {
    v16bf a;
#pragma unroll
    for (int e = 0; e < 16; ++e)
      a[e] = (__bf16)ctxs[kmap16(lane, e) * DH_ + i * 16 + (lane & 15)];
    v8f acc = {};
    acc = wmma_bf16(a, bm, acc);
    v8bf pk;
#pragma unroll
    for (int r = 0; r < 8; ++r) pk[r] = (__bf16)acc[r];
    *(v8bf*)(attn_t + ((size_t)b * N_ + n) * INNER_ + h * DH_ + i * 16 + off) = pk;
  }
}

// ---------------------------------------------------------------------------
// 6) out2[o][n] = sum_c wout[o][c] attn_t[n][c] + bout[o] -> d_out (unnormalized)
//    grid (64, 8, 16), block 256.
// ---------------------------------------------------------------------------
__global__ __launch_bounds__(256) void k_proj(const __bf16* __restrict__ attn_t,
                                              const __bf16* __restrict__ wob,
                                              const float* __restrict__ bout,
                                              float* __restrict__ out) {
  const int lane = threadIdx.x & 31;
  const int wave = threadIdx.x >> 5;
  const int b = blockIdx.z;
  const int oBase = blockIdx.y * 32 + (wave >> 2) * 16;
  const int nBase = blockIdx.x * 64 + (wave & 3) * 16;
  const int row = oBase + (lane & 15);
  const int col = nBase + (lane & 15);
  const int off = (lane >> 4) << 3;
  const __bf16* bbase = attn_t + ((size_t)b * N_ + col) * INNER_;
  v8f acc = {};
#pragma unroll
  for (int k0 = 0; k0 < INNER_; k0 += 32) {
    v16bf a = frag_ld(wob + row * INNER_ + k0 + off);
    v16bf bm = frag_ld(bbase + k0 + off);
    acc = wmma_bf16(a, bm, acc);
  }
  float* ob = out + (size_t)b * C_ * N_;
#pragma unroll
  for (int r = 0; r < 8; ++r) {
    const int m = oBase + r + ((lane >> 4) << 3);
    ob[(size_t)m * N_ + col] = acc[r] + bout[m];
  }
}

// ---------------------------------------------------------------------------
// 7) per-batch partial sums (sum, sumsq).  grid (256, 16), block 256.
// ---------------------------------------------------------------------------
__global__ __launch_bounds__(256) void k_gn_partial(const float* __restrict__ out,
                                                    float* __restrict__ part) {
  const int b = blockIdx.y;
  const float* p = out + (size_t)b * C_ * N_ + (size_t)blockIdx.x * 4096;
  float s = 0.f, sq = 0.f;
  for (int i = threadIdx.x; i < 4096; i += 256) {
    const float v = p[i];
    s += v; sq += v * v;
  }
  __shared__ float ss[256], qq[256];
  const int t = threadIdx.x;
  ss[t] = s; qq[t] = sq; __syncthreads();
  for (int o = 128; o > 0; o >>= 1) {
    if (t < o) { ss[t] += ss[t + o]; qq[t] += qq[t + o]; }
    __syncthreads();
  }
  if (t == 0) {
    part[(size_t)(b * 256 + blockIdx.x) * 2 + 0] = ss[0];
    part[(size_t)(b * 256 + blockIdx.x) * 2 + 1] = qq[0];
  }
}

// ---------------------------------------------------------------------------
// 8) finalize mean/var + normalize + affine.  grid (256, 16), block 256.
// ---------------------------------------------------------------------------
__global__ __launch_bounds__(256) void k_gn_apply(float* __restrict__ out,
                                                  const float* __restrict__ part,
                                                  const float* __restrict__ gamma,
                                                  const float* __restrict__ beta) {
  const int b = blockIdx.y;
  const int t = threadIdx.x;
  __shared__ float ss[256], qq[256];
  ss[t] = part[(size_t)(b * 256 + t) * 2 + 0];
  qq[t] = part[(size_t)(b * 256 + t) * 2 + 1];
  __syncthreads();
  for (int o = 128; o > 0; o >>= 1) {
    if (t < o) { ss[t] += ss[t + o]; qq[t] += qq[t + o]; }
    __syncthreads();
  }
  const float invM = 1.0f / (float)(C_ * N_);
  const float mean = ss[0] * invM;
  const float var  = qq[0] * invM - mean * mean;
  const float inv  = rsqrtf(var + EPS_);
  float* p = out + (size_t)b * C_ * N_ + (size_t)blockIdx.x * 4096;
  for (int i = t; i < 4096; i += 256) {
    const int flat = blockIdx.x * 4096 + i;
    const int c = flat >> 12;
    p[i] = (p[i] - mean) * inv * gamma[c] + beta[c];
  }
}

// ---------------------------------------------------------------------------
extern "C" void kernel_launch(void* const* d_in, const int* in_sizes, int n_in,
                              void* d_out, int out_size, void* d_ws, size_t ws_size,
                              hipStream_t stream) {
  const float* x     = (const float*)d_in[0];
  const float* wqkv  = (const float*)d_in[1];
  const float* wout  = (const float*)d_in[2];
  const float* bout  = (const float*)d_in[3];
  const float* gamma = (const float*)d_in[4];
  const float* beta  = (const float*)d_in[5];
  float* out = (float*)d_out;
  (void)ws_size; (void)in_sizes; (void)n_in; (void)out_size;

  const size_t QSZ = (size_t)B_ * INNER_ * N_;   // 8,388,608
  float*  qf  = (float*)d_ws;                    // fp32 q (later reused as attn_t)
  float*  kf  = qf + QSZ;                        // fp32 k (bf16 k' in place)
  __bf16* vbf = (__bf16*)(kf + QSZ);             // bf16 v
  __bf16* qbt = vbf + QSZ;                       // bf16 q_t [bh][n][32]
  __bf16* wqb = qbt + QSZ;                       // bf16 w_qkv
  __bf16* wob = wqb + (size_t)O3_ * C_;          // bf16 w_out
  float*  ctx = (float*)(wob + (size_t)C_ * INNER_);
  float*  part = ctx + (size_t)B_ * HEADS_ * DH_ * DH_;
  __bf16* attn_t = (__bf16*)qf;                  // reuse dead q region

  k_cvt_w    <<<dim3(512),        256, 0, stream>>>(wqkv, wout, wqb, wob);
  k_qkv_gemm <<<dim3(64, 12, 16), 256, 0, stream>>>(x, wqb, qf, kf, vbf);
  k_ksoftmax <<<dim3(32, 4, 16),  256, 0, stream>>>(kf);
  k_qsoftmax <<<dim3(16, 4, 16),  256, 0, stream>>>(qf, qbt);
  k_context  <<<dim3(64),         256, 0, stream>>>(kf, vbf, ctx);
  k_attn_out <<<dim3(32, 64),     256, 0, stream>>>(qbt, ctx, attn_t);
  k_proj     <<<dim3(64, 8, 16),  256, 0, stream>>>(attn_t, wob, bout, out);
  k_gn_partial<<<dim3(256, 16),   256, 0, stream>>>(out, part);
  k_gn_apply <<<dim3(256, 16),    256, 0, stream>>>(out, part, gamma, beta);
}
